// BinaryConv1d_77249281786108
// MI455X (gfx1250) — 8-bit matrix operand order repaired and checked by reading, compiled, not run on hardware
//
#include <hip/hip_runtime.h>
#include <stdint.h>

#define B_     8
#define CIN    256
#define T_     16000
#define COUT   512
#define KW     3
#define KDIM   (CIN*KW)
#define NKSTEP 12
#define MTILES (COUT/16)
#define NTILES (T_/16)

typedef int v8i __attribute__((ext_vector_type(8)));
typedef int v4i __attribute__((ext_vector_type(4)));

__global__ void k_pad_zero(uint8_t* __restrict__ xsT) {
    int i = threadIdx.x;
    if (i >= 64) return;
    for (int b = 0; b < B_; ++b) {
        uint32_t* r0 = (uint32_t*)(xsT + (size_t)b * (T_ + 2) * CIN);
        uint32_t* r1 = (uint32_t*)(xsT + ((size_t)b * (T_ + 2) + T_ + 1) * CIN);
        r0[i] = 0u;
        r1[i] = 0u;
    }
}

__global__ void k_binx(const float* __restrict__ x, uint8_t* __restrict__ xsT) {
    __shared__ int8_t s[32][33];
    int tx = threadIdx.x & 31;
    int tg = threadIdx.x >> 5;
    int t0  = blockIdx.x * 32;
    int ci0 = blockIdx.y * 32;
    int b   = blockIdx.z;
    const float* xb = x + (size_t)b * CIN * T_;
#pragma unroll
    for (int i = 0; i < 4; ++i) {
        int ci = tg + i * 8;
        float v = xb[(size_t)(ci0 + ci) * T_ + t0 + tx];
        s[ci][tx] = (v >= 0.0f) ? (int8_t)1 : (int8_t)-1;
    }
    __syncthreads();
    int r  = threadIdx.x >> 3;
    int cd = threadIdx.x & 7;
    uint32_t wd =  (uint32_t)(uint8_t)s[cd * 4 + 0][r]
                | ((uint32_t)(uint8_t)s[cd * 4 + 1][r] << 8)
                | ((uint32_t)(uint8_t)s[cd * 4 + 2][r] << 16)
                | ((uint32_t)(uint8_t)s[cd * 4 + 3][r] << 24);
    size_t row = (size_t)b * (T_ + 2) + (t0 + r + 1);
    *(uint32_t*)(xsT + row * CIN + ci0 + cd * 4) = wd;
}

__global__ void k_alpha(const float* __restrict__ w, float* __restrict__ alpha) {
    int co = blockIdx.x * blockDim.x + threadIdx.x;
    if (co >= COUT) return;
    const float* wp = w + (size_t)co * KDIM;
    float s = 0.0f;
    for (int i = 0; i < KDIM; ++i) s += fabsf(wp[i]);
    alpha[co] = s * (1.0f / (float)KDIM);
}

__global__ void k_beta_p(const float* __restrict__ x, float* __restrict__ part) {
    __shared__ float sm[256];
    size_t base = (size_t)blockIdx.x * 4096;
    float s = 0.0f;
    for (int i = threadIdx.x; i < 4096; i += 256) s += fabsf(x[base + i]);
    sm[threadIdx.x] = s;
    __syncthreads();
    for (int off = 128; off > 0; off >>= 1) {
        if (threadIdx.x < off) sm[threadIdx.x] += sm[threadIdx.x + off];
        __syncthreads();
    }
    if (threadIdx.x == 0) part[blockIdx.x] = sm[0];
}

__global__ void k_beta_f(const float* __restrict__ part, float* __restrict__ beta) {
    __shared__ float sm[256];
    int b = blockIdx.x;
    float s = 0.0f;
    for (int i = threadIdx.x; i < 1000; i += 256) s += part[b * 1000 + i];
    sm[threadIdx.x] = s;
    __syncthreads();
    for (int off = 128; off > 0; off >>= 1) {
        if (threadIdx.x < off) sm[threadIdx.x] += sm[threadIdx.x + off];
        __syncthreads();
    }
    if (threadIdx.x == 0) beta[b] = sm[0] * (1.0f / ((float)CIN * (float)T_));
}

__global__ void k_packw(const float* __restrict__ w, uint32_t* __restrict__ apack) {
    int idx = blockIdx.x * 256 + threadIdx.x;
    int v     = idx & 7;
    int lane  = (idx >> 3) & 31;
    int kstep = (idx >> 8) % NKSTEP;
    int mt    = idx / (NKSTEP * 256);
    int m = lane & 15, half = lane >> 4;
    int co = mt * 16 + m;
    uint32_t word = 0;
#pragma unroll
    for (int j = 0; j < 4; ++j) {
        int K  = kstep * 64 + (v >> 2) * 32 + (v & 3) * 4 + half * 16 + j;
        int k  = K >> 8;
        int ci = K & 255;
        float wv = w[((size_t)co * CIN + ci) * KW + k];
        uint32_t sb = (wv >= 0.0f) ? 1u : 0xFFu;
        word |= sb << (8 * j);
    }
    apack[idx] = word;
}

__global__ void __launch_bounds__(256) k_gemm(const uint8_t* __restrict__ xsT,
                                              const uint32_t* __restrict__ apack,
                                              const float* __restrict__ alpha,
                                              const float* __restrict__ beta,
                                              float* __restrict__ out) {
    int lane = threadIdx.x & 31;
    int wid  = threadIdx.x >> 5;
    int tile = blockIdx.x * 8 + wid;
    int mt   = tile & 31;
    int rest = tile >> 5;
    int nt   = rest % NTILES;
    int b    = rest / NTILES;

    int N = lane & 15, halfB = lane >> 4;
    int t0 = nt * 16;

    const uint8_t* brow = xsT + ((size_t)b * (T_ + 2) + t0 + N) * CIN + halfB * 16;
    const uint32_t* ap  = apack + (size_t)mt * (NKSTEP * 256) + lane * 8;

    v8i acc = {0, 0, 0, 0, 0, 0, 0, 0};
#pragma unroll
    for (int kstep = 0; kstep < NKSTEP; ++kstep) {
        int k  = kstep >> 2;
        int cb = (kstep & 3) * 64;
        const uint8_t* p = brow + (size_t)k * CIN + cb;
        v4i blo = *(const v4i*)(p);
        v4i bhi = *(const v4i*)(p + 32);
        v8i bm;
        bm[0] = blo[0]; bm[1] = blo[1]; bm[2] = blo[2]; bm[3] = blo[3];
        bm[4] = bhi[0]; bm[5] = bhi[1]; bm[6] = bhi[2]; bm[7] = bhi[3];
        v8i am = *(const v8i*)(ap + kstep * 256);
        acc = __builtin_amdgcn_wmma_i32_16x16x64_iu8(true, am, true, bm, acc,
                                                     false, false);
    }

    float bs = beta[b];
    size_t obase = ((size_t)b * COUT + mt * 16) * (size_t)T_ + t0 + N;
#pragma unroll
    for (int r = 0; r < 8; ++r) {
        int M = r + halfB * 8;
        float a = alpha[mt * 16 + M];
        out[obase + (size_t)M * T_] = (float)acc[r] * a * bs;
    }
}

extern "C" void kernel_launch(void* const* d_in, const int* in_sizes, int n_in,
                              void* d_out, int out_size, void* d_ws, size_t ws_size,
                              hipStream_t stream) {
    const float* x = (const float*)d_in[0];
    const float* w = (const float*)d_in[1];
    float* out = (float*)d_out;

    uint8_t* ws = (uint8_t*)d_ws;
    float*    part  = (float*)ws;
    float*    beta  = (float*)(ws + 32768);
    float*    alpha = (float*)(ws + 33024);
    uint32_t* apack = (uint32_t*)(ws + 36864);
    uint8_t*  xsT   = ws + 430080;

    k_pad_zero<<<1, 256, 0, stream>>>(xsT);
    k_binx<<<dim3(T_ / 32, CIN / 32, B_), 256, 0, stream>>>(x, xsT);
    k_alpha<<<(COUT + 255) / 256, 256, 0, stream>>>(w, alpha);
    k_beta_p<<<8000, 256, 0, stream>>>(x, part);
    k_beta_f<<<B_, 256, 0, stream>>>(part, beta);
    k_packw<<<(MTILES * NKSTEP * 256) / 256, 256, 0, stream>>>(w, apack);
    k_gemm<<<(B_ * NTILES * MTILES) / 8, 256, 0, stream>>>(xsT, apack, alpha, beta, out);
}
